// QuantizedBasicBlock_37177236914843
// MI455X (gfx1250) — 8-bit matrix operand order repaired and checked by reading, compiled, not run on hardware
//
#include <hip/hip_runtime.h>


typedef __attribute__((ext_vector_type(8))) int v8i;
typedef __attribute__((ext_vector_type(4))) int i4v;

#define NBATCH 64
#define CCH    256
#define HH     28
#define WW     28
#define HP     30
#define WP     30
#define KTOT   2304
#define MROWS  (NBATCH*HH*WW)
#define KSTEPS 36

#if __has_builtin(__builtin_amdgcn_global_load_async_to_lds_b128) && \
    __has_builtin(__builtin_amdgcn_s_wait_asynccnt)
#define HAS_ASYNC_LDS 1
#else
#define HAS_ASYNC_LDS 0
#endif

#if HAS_ASYNC_LDS
__device__ __forceinline__ void async_copy_b128(const void* gsrc, void* ldst) {
  __builtin_amdgcn_global_load_async_to_lds_b128(
      (__attribute__((address_space(1))) i4v*)gsrc,
      (__attribute__((address_space(3))) i4v*)ldst, 0, 0);
}
#endif

union AFrag { uint2 u2[4]; v8i v; };
union BFrag { uint4 u4[2]; v8i v; };

__global__ void pack_x_kernel(const float* __restrict__ x,
                              const float* __restrict__ zp,
                              unsigned char* __restrict__ XP) {
  int idx = blockIdx.x * 256 + threadIdx.x;
  int c    = idx & 255;
  int rest = idx >> 8;
  int wp   = rest % WP; rest /= WP;
  int hp   = rest % HP;
  int n    = rest / HP;
  unsigned char q;
  if (hp >= 1 && hp <= HH && wp >= 1 && wp <= WW) {
    float v = x[(((size_t)n * CCH + c) * HH + (hp - 1)) * WW + (wp - 1)];
    q = (unsigned char)(int)v;
  } else {
    q = (unsigned char)(int)rintf(zp[0]);
  }
  XP[idx] = q;
}

__global__ void fill_pad_kernel(const float* __restrict__ zp,
                                unsigned char* __restrict__ YP) {
  int idx = blockIdx.x * 256 + threadIdx.x;
  YP[idx] = (unsigned char)(int)rintf(zp[0]);
}

__global__ void pack_w_kernel(const float* __restrict__ w,
                              signed char* __restrict__ Wt) {
  int idx = blockIdx.x * 256 + threadIdx.x;
  int k  = idx % KTOT;
  int co = idx / KTOT;
  int t  = k >> 8;
  int ci = k & 255;
  float v = w[((size_t)co * CCH + ci) * 9 + t];
  Wt[(size_t)co * KTOT + k] = (signed char)(int)v;
}

__global__ void wsum_kernel(const float* __restrict__ w,
                            int* __restrict__ wsum) {
  int co = threadIdx.x;
  const float* p = w + (size_t)co * KTOT;
  int s = 0;
  for (int j = 0; j < KTOT; ++j) s += (int)p[j];
  wsum[co] = s;
}

template <bool FUSE>
__global__ __launch_bounds__(256)
void qconv_wmma_kernel(const unsigned char* __restrict__ Ain,
                       const signed char*  __restrict__ Wt,
                       const int*          __restrict__ wsum,
                       const float* sp_in, const float* zp_in,
                       const float* sp_w,  const float* sp_out, const float* zp_out,
                       const unsigned char* __restrict__ XPin,
                       const float* sp_x, const float* zp_x,
                       const float* sp_add, const float* zp_add,
                       unsigned char* __restrict__ Yq,
                       float* __restrict__ Out)
{
  __shared__ unsigned char Bs[2][64 * 64];

  const int tid  = threadIdx.x;
  const int lane = tid & 31;
  const int wv   = tid >> 5;
  const int co0  = blockIdx.y << 6;
  const int m0   = (blockIdx.x << 8) + (wv << 5);

  int an[2], ah[2], aw[2];
#pragma unroll
  for (int mt = 0; mt < 2; ++mt) {
    const int arow = m0 + (mt << 4) + (lane & 15);
    an[mt] = arow / (HH * WW);
    const int ar = arow % (HH * WW);
    ah[mt] = ar / WW;
    aw[mt] = ar % WW;
  }

  const int sco   = tid >> 2;
  const int spart = tid & 3;
  const signed char* wrow = Wt + (size_t)(co0 + sco) * KTOT + spart * 16;
  unsigned char* sdst0 = &Bs[0][sco * 64 + spart * 16];
  unsigned char* sdst1 = &Bs[1][sco * 64 + spart * 16];

  v8i acc[2][4];
  const v8i vzero = {0, 0, 0, 0, 0, 0, 0, 0};
#pragma unroll
  for (int mt = 0; mt < 2; ++mt)
#pragma unroll
    for (int j = 0; j < 4; ++j) acc[mt][j] = vzero;

#if HAS_ASYNC_LDS
  async_copy_b128(wrow, sdst0);
  __builtin_amdgcn_s_wait_asynccnt(0);
#else
  *(uint4*)sdst0 = *(const uint4*)(wrow);
#endif
  __syncthreads();

#pragma unroll 1
  for (int kb = 0; kb < KSTEPS; ++kb) {
    const int buf = kb & 1;
    unsigned char* sdst_nxt = buf ? sdst0 : sdst1;

#if HAS_ASYNC_LDS
    if (kb + 1 < KSTEPS) async_copy_b128(wrow + (kb + 1) * 64, sdst_nxt);
#else
    uint4 pre;
    if (kb + 1 < KSTEPS) pre = *(const uint4*)(wrow + (kb + 1) * 64);
#endif

    const int t   = kb >> 2;
    const int ci0 = (kb & 3) << 6;
    const int ky  = t / 3;
    const int kx  = t % 3;
    AFrag a[2];
#pragma unroll
    for (int mt = 0; mt < 2; ++mt) {
      const unsigned char* ap =
          Ain + (((size_t)((an[mt] * HP + ah[mt] + ky) * WP + (aw[mt] + kx))) << 8)
              + ci0 + ((lane >> 4) << 4);
      a[mt].u2[0] = *(const uint2*)(ap +  0);
      a[mt].u2[1] = *(const uint2*)(ap +  8);
      a[mt].u2[2] = *(const uint2*)(ap + 32);
      a[mt].u2[3] = *(const uint2*)(ap + 40);
    }

#pragma unroll
    for (int jt = 0; jt < 4; ++jt) {
      const unsigned char* bp =
          &Bs[buf][(((jt << 4) + (lane & 15)) << 6) + ((lane >> 4) << 4)];
      BFrag b;
      b.u4[0] = *(const uint4*)(bp);
      b.u4[1] = *(const uint4*)(bp + 32);
#pragma unroll
      for (int mt = 0; mt < 2; ++mt) {
        acc[mt][jt] = __builtin_amdgcn_wmma_i32_16x16x64_iu8(
             false, a[mt].v,  true, b.v, acc[mt][jt],
             false,  false);
      }
    }

#if HAS_ASYNC_LDS
    if (kb + 1 < KSTEPS) __builtin_amdgcn_s_wait_asynccnt(0);
#else
    if (kb + 1 < KSTEPS) *(uint4*)sdst_nxt = pre;
#endif
    __syncthreads();
  }

  const float Ms   = sp_in[0] * sp_w[0] / sp_out[0];
  const float zin  = zp_in[0];
  const float zout = zp_out[0];
  float sxa = 0.f, zx = 0.f, soa = 0.f, zadd = 0.f;
  if (FUSE) {
    const float sad = sp_add[0];
    sxa  = sp_x[0] / sad;
    zx   = zp_x[0];
    soa  = sp_out[0] / sad;
    zadd = zp_add[0];
  }

#pragma unroll
  for (int mt = 0; mt < 2; ++mt) {
#pragma unroll
    for (int r = 0; r < 8; ++r) {
      const int mm = m0 + (mt << 4) + r + ((lane >> 4) << 3);
      const int n2 = mm / (HH * WW);
      const int r2 = mm % (HH * WW);
      const int h2 = r2 / WW;
      const int w2 = r2 % WW;
      const size_t offP = ((size_t)((n2 * HP + h2 + 1) * WP + (w2 + 1))) << 8;
#pragma unroll
      for (int jt = 0; jt < 4; ++jt) {
        const int co = co0 + (jt << 4) + (lane & 15);
        const float f = (float)acc[mt][jt][r] - zin * (float)wsum[co];
        float q = rintf(Ms * f) + zout;
        q = fminf(fmaxf(q, 0.0f), 255.0f);
        if (FUSE) {
          const float xv = (float)XPin[offP + co];
          const float y  = sxa * (xv - zx) + soa * (q - zout);
          float o = rintf(y) + zadd;
          o = fminf(fmaxf(o, 0.0f), 255.0f);
          Out[((size_t)n2 * CCH + co) * (HH * WW) + h2 * WW + w2] = o;
        } else {
          Yq[offP + co] = (unsigned char)q;
        }
      }
    }
  }
}

extern "C" void kernel_launch(void* const* d_in, const int* in_sizes, int n_in,
                              void* d_out, int out_size, void* d_ws, size_t ws_size,
                              hipStream_t stream) {
  (void)in_sizes; (void)n_in; (void)out_size; (void)ws_size;
  const float* x    = (const float*)d_in[0];
  const float* w1   = (const float*)d_in[1];
  const float* w2   = (const float*)d_in[2];
  const float* s_x  = (const float*)d_in[3];
  const float* z_x  = (const float*)d_in[4];
  const float* s_w1 = (const float*)d_in[5];
  const float* s_c1 = (const float*)d_in[6];
  const float* z_c1 = (const float*)d_in[7];
  const float* s_w2 = (const float*)d_in[8];
  const float* s_c2 = (const float*)d_in[9];
  const float* z_c2 = (const float*)d_in[10];
  const float* s_ad = (const float*)d_in[11];
  const float* z_ad = (const float*)d_in[12];
  float* out = (float*)d_out;

  unsigned char* ws8 = (unsigned char*)d_ws;
  const size_t PADSZ = (size_t)NBATCH * HP * WP * CCH;
  const size_t WTSZ  = (size_t)CCH * KTOT;
  unsigned char* XP  = ws8;
  unsigned char* YP  = ws8 + PADSZ;
  signed char*   W1t = (signed char*)(ws8 + 2 * PADSZ);
  signed char*   W2t = (signed char*)(ws8 + 2 * PADSZ + WTSZ);
  int*           S1  = (int*)(ws8 + 2 * PADSZ + 2 * WTSZ);
  int*           S2  = S1 + CCH;

  pack_x_kernel  <<<(unsigned)(PADSZ / 256), 256, 0, stream>>>(x, z_x, XP);
  fill_pad_kernel<<<(unsigned)(PADSZ / 256), 256, 0, stream>>>(z_c1, YP);
  pack_w_kernel  <<<(unsigned)(WTSZ / 256), 256, 0, stream>>>(w1, W1t);
  pack_w_kernel  <<<(unsigned)(WTSZ / 256), 256, 0, stream>>>(w2, W2t);
  wsum_kernel    <<<1, 256, 0, stream>>>(w1, S1);
  wsum_kernel    <<<1, 256, 0, stream>>>(w2, S2);

  dim3 grid(MROWS / 256, 4);
  qconv_wmma_kernel<false><<<grid, 256, 0, stream>>>(
      XP, W1t, S1, s_x, z_x, s_w1, s_c1, z_c1,
      nullptr, nullptr, nullptr, nullptr, nullptr, YP, nullptr);
  qconv_wmma_kernel<true><<<grid, 256, 0, stream>>>(
      YP, W2t, S2, s_c1, z_c1, s_w2, s_c2, z_c2,
      XP, s_x, z_x, s_ad, z_ad, nullptr, out);
}
